// PxByEmbed_13374528159749
// MI455X (gfx1250) — matrix operand order repaired and checked by reading, not compiled, not run on hardware
//
#include <hip/hip_runtime.h>
#include <stdint.h>

typedef __attribute__((ext_vector_type(16))) __bf16 bf16x16;
typedef __attribute__((ext_vector_type(8)))  float  f32x8;

#define DIMM   512
#define CIN    504
#define KPAD   512
#define EMODEL 56
#define NROWS  65536
#define LN_EPS 1e-5f

__device__ __forceinline__ unsigned short f2bf(float f) {
    union { float f; uint32_t u; } c; c.f = f;
    uint32_t u = c.u;
    uint32_t r = (u + 0x7FFFu + ((u >> 16) & 1u)) >> 16;  // round-to-nearest-even
    return (unsigned short)r;
}

// ---------------- Kernel 0: embedding table f32 -> bf16 ----------------
__global__ void k_emb_bf16(const float* __restrict__ emb,
                           unsigned short* __restrict__ out, int n) {
    int i = blockIdx.x * blockDim.x + threadIdx.x;
    if (i < n) out[i] = f2bf(emb[i]);
}

// ---------------- Kernel 1: fold conv+gate+projection into W_eff^T, b_eff ----
// wT layout: [N=512][Kpad=512] bf16 (N-major), K in [504,512) zero-padded.
__global__ void k_build_weff(const float* __restrict__ alpha,
                             const float* __restrict__ conv_w,
                             const float* __restrict__ conv_b,
                             const float* __restrict__ proj_w,
                             const float* __restrict__ proj_b,
                             unsigned short* __restrict__ wT,
                             float* __restrict__ beff) {
    int tid = blockIdx.x * blockDim.x + threadIdx.x;   // 0 .. 512*512-1
    int n = tid >> 9;
    int k = tid & 511;
    float g[9];
#pragma unroll
    for (int p = 0; p < 9; ++p) g[p] = 1.0f / (1.0f + __expf(-alpha[p]));
    if (k < CIN) {
        int e   = k / 9;
        int pin = k - e * 9;
        int ii = pin / 3, jj = pin - ii * 3;
        float val = g[pin] * proj_w[n * CIN + k];
#pragma unroll
        for (int po = 0; po < 9; ++po) {
            int io = po / 3, jo = po - io * 3;
            int di = ii - io + 1, dj = jj - jo + 1;
            if ((unsigned)di > 2u || (unsigned)dj > 2u) continue;
            const float* pw = proj_w + n * CIN + po;          // stride 9 over o
            const float* cw = conv_w + e * 9 + di * 3 + dj;   // stride 504 over o
            float s = 0.f;
            for (int o = 0; o < EMODEL; ++o) s += pw[o * 9] * cw[o * 504];
            val += (1.0f - g[po]) * s;
        }
        wT[n * KPAD + k] = f2bf(val);
    } else {
        wT[n * KPAD + k] = 0;
        if (k == CIN) {
            float b = proj_b[n];
            for (int po = 0; po < 9; ++po) {
                float w1 = 1.0f - g[po];
                for (int o = 0; o < EMODEL; ++o)
                    b += proj_w[n * CIN + o * 9 + po] * w1 * conv_b[o];
            }
            beff[n] = b;
        }
    }
}

// ---------------- Kernel 2: fused gather + WMMA GEMM + bias + LayerNorm ------
__global__ __launch_bounds__(256)
void k_gemm_ln(const int* __restrict__ x,
               const unsigned short* __restrict__ embb,
               const unsigned short* __restrict__ wT,
               const float* __restrict__ beff,
               const float* __restrict__ gamma,
               const float* __restrict__ beta,
               float* __restrict__ out) {
    __shared__ unsigned short As[32 * KPAD];   // 32 rows x 512 K, bf16 (32 KB)
    __shared__ int   xi[32 * 9];
    __shared__ float redS[32];
    __shared__ float redQ[32];

    const int tid = threadIdx.x;
    const int r0  = blockIdx.x * 32;

    if (tid < 32) { redS[tid] = 0.f; redQ[tid] = 0.f; }
    if (tid < 288) xi[tid] = x[r0 * 9 + tid];
    __syncthreads();

    // Gather embedded rows into LDS (channel-major c = e*9+pos, zero pad K>=504)
    for (int s = tid; s < 32 * KPAD; s += 256) {
        int r = s >> 9;
        int c = s & 511;
        unsigned short v = 0;
        if (c < CIN) {
            int e = c / 9;
            int p = c - e * 9;
            v = embb[xi[r * 9 + p] * EMODEL + e];
        }
        As[s] = v;
    }
    __syncthreads();

    const int wave  = tid >> 5;
    const int lane  = tid & 31;
    const int half  = lane >> 4;   // 0|1
    const int lcol  = lane & 15;
    const int nbase = wave * 64;

    f32x8 acc[2][4];
#pragma unroll
    for (int mt = 0; mt < 2; ++mt)
#pragma unroll
        for (int nt = 0; nt < 4; ++nt)
#pragma unroll
            for (int e = 0; e < 8; ++e) acc[mt][nt][e] = 0.f;

    union Frag { bf16x16 v; uint4 q[2]; };

#pragma unroll 4
    for (int kt = 0; kt < 16; ++kt) {
        Frag a[2];
#pragma unroll
        for (int mt = 0; mt < 2; ++mt) {
            int base = (mt * 16 + lcol) * KPAD + kt * 32 + half * 8;
            a[mt].q[0] = *(const uint4*)(&As[base]);
            a[mt].q[1] = *(const uint4*)(&As[base + 16]);
        }
#pragma unroll
        for (int nt = 0; nt < 4; ++nt) {
            Frag b;
            int ncol = nbase + nt * 16 + lcol;
            const uint4* bp = (const uint4*)(wT + ncol * KPAD + kt * 32 + half * 8);
            b.q[0] = bp[0];
            b.q[1] = bp[2];
            acc[0][nt] = __builtin_amdgcn_wmma_f32_16x16x32_bf16(
                false, a[0].v, false, b.v, (short)0, acc[0][nt], false, false);
            acc[1][nt] = __builtin_amdgcn_wmma_f32_16x16x32_bf16(
                false, a[1].v, false, b.v, (short)0, acc[1][nt], false, false);
        }
    }

    // Bias + LayerNorm partials
    float be[4], gm[4], bt[4];
#pragma unroll
    for (int nt = 0; nt < 4; ++nt) {
        int ncol = nbase + nt * 16 + lcol;
        be[nt] = beff[ncol];
        gm[nt] = gamma[ncol];
        bt[nt] = beta[ncol];
    }
#pragma unroll
    for (int mt = 0; mt < 2; ++mt) {
#pragma unroll
        for (int r = 0; r < 8; ++r) {
            float ps = 0.f, pq = 0.f;
#pragma unroll
            for (int nt = 0; nt < 4; ++nt) {
                float y = acc[mt][nt][r] + be[nt];
                acc[mt][nt][r] = y;
                ps += y;
                pq += y * y;
            }
#pragma unroll
            for (int off = 1; off < 16; off <<= 1) {
                ps += __shfl_xor(ps, off, 32);
                pq += __shfl_xor(pq, off, 32);
            }
            if (lcol == 0) {
                int m = mt * 16 + half * 8 + r;
                atomicAdd(&redS[m], ps);
                atomicAdd(&redQ[m], pq);
            }
        }
    }
    __syncthreads();
    if (tid < 32) {
        float mu  = redS[tid] * (1.0f / 512.0f);
        float var = redQ[tid] * (1.0f / 512.0f) - mu * mu;
        redS[tid] = mu;
        redQ[tid] = rsqrtf(var + LN_EPS);
    }
    __syncthreads();

#pragma unroll
    for (int mt = 0; mt < 2; ++mt) {
#pragma unroll
        for (int r = 0; r < 8; ++r) {
            int m = mt * 16 + half * 8 + r;
            float mu = redS[m];
            float rs = redQ[m];
#pragma unroll
            for (int nt = 0; nt < 4; ++nt) {
                int ncol = nbase + nt * 16 + lcol;
                out[(size_t)(r0 + m) * DIMM + ncol] =
                    (acc[mt][nt][r] - mu) * rs * gm[nt] + bt[nt];
            }
        }
    }
}

extern "C" void kernel_launch(void* const* d_in, const int* in_sizes, int n_in,
                              void* d_out, int out_size, void* d_ws, size_t ws_size,
                              hipStream_t stream) {
    (void)in_sizes; (void)n_in; (void)out_size; (void)ws_size;
    const int*   x      = (const int*)d_in[0];
    const float* embf   = (const float*)d_in[1];
    const float* alpha  = (const float*)d_in[2];
    const float* conv_w = (const float*)d_in[3];
    const float* conv_b = (const float*)d_in[4];
    const float* proj_w = (const float*)d_in[5];
    const float* proj_b = (const float*)d_in[6];
    const float* gamma  = (const float*)d_in[7];
    const float* beta   = (const float*)d_in[8];
    float* out = (float*)d_out;

    char* ws = (char*)d_ws;
    unsigned short* wT   = (unsigned short*)(ws);                          // 512 KB
    float*          beff = (float*)(ws + 512 * 512 * 2);                   // 2 KB
    unsigned short* embb = (unsigned short*)(ws + 512 * 512 * 2 + 512 * 4); // 28 KB

    k_emb_bf16 <<<(256 * EMODEL + 255) / 256, 256, 0, stream>>>(embf, embb, 256 * EMODEL);
    k_build_weff<<<(512 * 512) / 256, 256, 0, stream>>>(alpha, conv_w, conv_b,
                                                        proj_w, proj_b, wT, beff);
    k_gemm_ln  <<<NROWS / 32, 256, 0, stream>>>(x, embb, wT, beff, gamma, beta, out);
}
